// EnhancedMemoryEfficientAttention_21887153341134
// MI455X (gfx1250) — compile-verified, matrix operand order checked by reading, not run on hardware
//
#include <hip/hip_runtime.h>
#include <hip/hip_bf16.h>
#include <stdint.h>

typedef __attribute__((ext_vector_type(16))) _Float16 v16h;
typedef __attribute__((ext_vector_type(2)))  __fp16   v2h;   // matches cvt_pkrtz return
typedef __attribute__((ext_vector_type(8)))  float    v8f;
typedef __attribute__((ext_vector_type(4)))  float    v4f;

#define HEADS  4
#define DH     32
#define CHUNK  2048
#define NC     8
#define HW     16384
#define SCALE  0.17677669529663687f     // 32^-0.5
#define GN_EPS 1e-5f
#define KSTAGE 512                      // k rows staged in LDS per async round

// ---- WMMA helper: D = A(16x32 f16) * B(32x16 f16) + C(f32) ----
__device__ __forceinline__ v8f wmma16(v16h a, v16h b, v8f c) {
  return __builtin_amdgcn_wmma_f32_16x16x32_f16(false, a, false, b, (short)0, c,
                                                false, false);
}

// Load a 16-half A/B operand for one lane from a 32-element K-row.
// Per ISA 7.12.2 (16-bit A 16x32): lane-half h holds K = 8h..8h+7 (VGPR0-3)
// and K = 16+8h..16+8h+7 (VGPR4-7).  16B-aligned chunks.
__device__ __forceinline__ v16h load16(const _Float16* row, int h) {
  union { v16h v; uint4 u[2]; } t;
  t.u[0] = *(const uint4*)(row + 8 * h);
  t.u[1] = *(const uint4*)(row + 16 + 8 * h);
  return t.v;
}

// Same operand layout but converting from f32 memory (weights).
__device__ __forceinline__ v16h cvt_a(const float* row, int h) {
  union { v16h v; v2h h2[8]; } t;
  const float* p0 = row + 8 * h;
  const float* p1 = row + 16 + 8 * h;
#pragma unroll
  for (int i = 0; i < 4; ++i) t.h2[i]     = __builtin_amdgcn_cvt_pkrtz(p0[2*i], p0[2*i+1]);
#pragma unroll
  for (int i = 0; i < 4; ++i) t.h2[4 + i] = __builtin_amdgcn_cvt_pkrtz(p1[2*i], p1[2*i+1]);
  return t.v;
}

__device__ __forceinline__ uint4 pack8(const float* v) {
  union { v2h h2[4]; uint4 u; } t;
#pragma unroll
  for (int i = 0; i < 4; ++i) t.h2[i] = __builtin_amdgcn_cvt_pkrtz(v[2*i], v[2*i+1]);
  return t.u;
}

// gfx1250 async global->LDS copy of 16 bytes (tracked by ASYNCcnt).
__device__ __forceinline__ void async_b128(unsigned lds_off, const void* gaddr) {
  asm volatile("global_load_async_to_lds_b128 %0, %1, off"
               :: "v"(lds_off), "v"(gaddr) : "memory");
}
__device__ __forceinline__ void wait_async0() {
  asm volatile("s_wait_asynccnt 0x0" ::: "memory");
}

// ===================== Kernel 1: QKV projection (WMMA) =====================
// qkv[o,p] = sum_c w_qkv[o,c]*x[c,p];  o<128->q, <256->k, else v (f16 outputs)
__global__ __launch_bounds__(256) void k_qkv(const float* __restrict__ x,
                                             const float* __restrict__ wq,
                                             _Float16* __restrict__ qh,
                                             _Float16* __restrict__ kh,
                                             _Float16* __restrict__ vh) {
  __shared__ _Float16 xT[128 * 128];  // [p][c] f16, 32 KB
  const int lane = threadIdx.x & 31, wave = threadIdx.x >> 5;
  const int h = lane >> 4, lm = lane & 15;
  const int pbase = blockIdx.x * 128;

  for (int c = wave; c < 128; c += 8) {
    v4f xv = *(const v4f*)(x + (size_t)c * HW + pbase + lane * 4);
#pragma unroll
    for (int u = 0; u < 4; ++u)
      xT[(lane * 4 + u) * 128 + c] = (_Float16)xv[u];
  }
  __syncthreads();

  for (int ot = wave; ot < 24; ot += 8) {
    const int obase = ot * 16;
    v16h aT[4];
#pragma unroll
    for (int cb = 0; cb < 4; ++cb)
      aT[cb] = cvt_a(wq + (size_t)(obase + lm) * 128 + cb * 32, h);
    for (int pj = 0; pj < 8; ++pj) {
      v8f acc = {0, 0, 0, 0, 0, 0, 0, 0};
#pragma unroll
      for (int cb = 0; cb < 4; ++cb) {
        v16h b = load16(xT + (pj * 16 + lm) * 128 + cb * 32, h);
        acc = wmma16(aT[cb], b, acc);
      }
      const int p = pbase + pj * 16 + lm;
      const int n = p >> 11, j = p & 2047;
      float tv[8];
#pragma unroll
      for (int r = 0; r < 8; ++r) tv[r] = acc[r];
      // branch is uniform per 16-row tile (obase), not per element
      if (obase < 128) {                 // Q: 8 consecutive d -> one b128
        const int head = obase >> 5, dbase = (obase & 31) + 8 * h;
        *(uint4*)(qh + (size_t)((head * NC + n) * CHUNK + j) * DH + dbase) = pack8(tv);
      } else if (obase < 256) {          // K: same contiguous-d layout
        const int o2 = obase - 128;
        const int head = o2 >> 5, dbase = (o2 & 31) + 8 * h;
        *(uint4*)(kh + (size_t)((head * NC + n) * CHUNK + j) * DH + dbase) = pack8(tv);
      } else {                           // V: [d][c] layout -> strided b16
        const int o3 = obase - 256;
        const int head = o3 >> 5, dbase = (o3 & 31) + 8 * h;
#pragma unroll
        for (int r = 0; r < 8; ++r)
          vh[(size_t)((head * NC + n) * DH + dbase + r) * CHUNK + j] = (_Float16)tv[r];
      }
    }
  }
}

// ============ Kernel 2: softmax row stats -> (max, 1/Z) pairs ==============
// K chunk is staged through LDS with gfx1250 async global->LDS copies.
__global__ __launch_bounds__(256) void k_rowstats(const _Float16* __restrict__ qh,
                                                  const _Float16* __restrict__ kh,
                                                  float* __restrict__ mz) {
  __shared__ _Float16 ksm[KSTAGE * DH];  // 32 KB staging buffer
  const int lane = threadIdx.x & 31, wave = threadIdx.x >> 5;
  const int h = lane >> 4, lm = lane & 15;
  const int hc = blockIdx.x >> 4;
  const int cbase = (blockIdx.x & 15) * 128 + wave * 16;
  const size_t base = (size_t)hc * CHUNK;
  const v8f vz = {0, 0, 0, 0, 0, 0, 0, 0};

  const v16h a = load16(qh + (base + cbase + lm) * DH, h);
  const unsigned lbase = (unsigned)(size_t)(&ksm[0]);

  v8f m8, z8 = vz;
#pragma unroll
  for (int r = 0; r < 8; ++r) m8[r] = -3.0e38f;

  for (int pass = 0; pass < 2; ++pass) {
    for (int e0 = 0; e0 < CHUNK; e0 += KSTAGE) {
      const char* gsrc = (const char*)(kh + (base + e0) * DH);
#pragma unroll
      for (int i = 0; i < (KSTAGE * DH * 2) / (256 * 16); ++i) {
        const unsigned off = (threadIdx.x + 256 * i) * 16;
        async_b128(lbase + off, gsrc + off);
      }
      wait_async0();
      __syncthreads();
      if (pass == 0) {
        for (int e = 0; e < KSTAGE; e += 16) {
          v16h b = load16(ksm + (e + lm) * DH, h);
          v8f s = wmma16(a, b, vz);
#pragma unroll
          for (int r = 0; r < 8; ++r) m8[r] = fmaxf(m8[r], s[r] * SCALE);
        }
      } else {
        for (int e = 0; e < KSTAGE; e += 16) {
          v16h b = load16(ksm + (e + lm) * DH, h);
          v8f s = wmma16(a, b, vz);
#pragma unroll
          for (int r = 0; r < 8; ++r) z8[r] += __expf(s[r] * SCALE - m8[r]);
        }
      }
      __syncthreads();  // protect staging buffer before next async round
    }
    if (pass == 0) {
#pragma unroll
      for (int r = 0; r < 8; ++r) {
        float v = m8[r];
        v = fmaxf(v, __shfl_xor(v, 1, 32));
        v = fmaxf(v, __shfl_xor(v, 2, 32));
        v = fmaxf(v, __shfl_xor(v, 4, 32));
        v = fmaxf(v, __shfl_xor(v, 8, 32));
        m8[r] = v;
      }
    }
  }
#pragma unroll
  for (int r = 0; r < 8; ++r) {
    float v = z8[r];
    v += __shfl_xor(v, 1, 32);
    v += __shfl_xor(v, 2, 32);
    v += __shfl_xor(v, 4, 32);
    v += __shfl_xor(v, 8, 32);
    z8[r] = v;
  }
  if (lm == 0) {
#pragma unroll
    for (int r = 0; r < 8; ++r) {
      const int row = cbase + r + 8 * h;
      float2 t;
      t.x = m8[r];
      t.y = 1.0f / z8[r];
      *(float2*)(mz + 2 * (base + row)) = t;
    }
  }
}

// ====== Kernel 3: out = V * softmax(QK^T), recompute S, double WMMA chain ==
__global__ __launch_bounds__(256) void k_attnout(const _Float16* __restrict__ qh,
                                                 const _Float16* __restrict__ kh,
                                                 const _Float16* __restrict__ vh,
                                                 const float* __restrict__ mz,
                                                 float* __restrict__ ao) {
  __shared__ _Float16 plds[8][512];  // per-wave 16j x 32c f16 staging (1 KB)
  const int lane = threadIdx.x & 31, wave = threadIdx.x >> 5;
  const int h = lane >> 4, lm = lane & 15;
  const int hc = blockIdx.x >> 4;
  const int head = hc >> 3, n = hc & 7;
  const int jb = (blockIdx.x & 15) * 128 + wave * 16;
  const size_t base = (size_t)hc * CHUNK;
  const v8f vz = {0, 0, 0, 0, 0, 0, 0, 0};

  const v16h bk = load16(kh + (base + jb + lm) * DH, h);  // K columns, resident
  v8f acc0 = vz, acc1 = vz;
  _Float16* myp = plds[wave];

  for (int c = 0; c < CHUNK; c += 32) {
    __builtin_prefetch(qh + (base + c + 64 + lm) * DH, 0, 0);
    v16h aq0 = load16(qh + (base + c + lm) * DH, h);
    v16h aq1 = load16(qh + (base + c + 16 + lm) * DH, h);
    v8f s0 = wmma16(aq0, bk, vz);
    v8f s1 = wmma16(aq1, bk, vz);

    float p0[8], p1[8];
#pragma unroll
    for (int r = 0; r < 8; ++r) {
      const int row0 = c + r + 8 * h, row1 = row0 + 16;
      const float2 t0 = *(const float2*)(mz + 2 * (base + row0));
      const float2 t1 = *(const float2*)(mz + 2 * (base + row1));
      p0[r] = __expf(s0[r] * SCALE - t0.x) * t0.y;
      p1[r] = __expf(s1[r] * SCALE - t1.x) * t1.y;
    }
    // stage P into B-operand layout [j][c] (per-wave private region)
    *(uint4*)(myp + lm * 32 + 8 * h)      = pack8(p0);
    *(uint4*)(myp + lm * 32 + 16 + 8 * h) = pack8(p1);

    v16h bp  = load16(myp + lm * 32, h);
    v16h av0 = load16(vh + ((size_t)hc * DH + lm) * CHUNK + c, h);
    v16h av1 = load16(vh + ((size_t)hc * DH + 16 + lm) * CHUNK + c, h);
    acc0 = wmma16(av0, bp, acc0);
    acc1 = wmma16(av1, bp, acc1);
  }

  const int p = n * CHUNK + jb + lm;
#pragma unroll
  for (int r = 0; r < 8; ++r) {
    const int d0 = r + 8 * h;
    ao[(size_t)(head * DH + d0) * HW + p]      = acc0[r];
    ao[(size_t)(head * DH + 16 + d0) * HW + p] = acc1[r];
  }
}

// ===================== Kernel 4: per-channel spatial mean ==================
__global__ __launch_bounds__(256) void k_chmean(const float* __restrict__ ao,
                                                float* __restrict__ mean) {
  const int ch = blockIdx.x;
  float s = 0.f;
  for (int i = threadIdx.x; i < HW; i += 256) s += ao[(size_t)ch * HW + i];
#pragma unroll
  for (int off = 16; off; off >>= 1) s += __shfl_xor(s, off, 32);
  __shared__ float red[8];
  if ((threadIdx.x & 31) == 0) red[threadIdx.x >> 5] = s;
  __syncthreads();
  if (threadIdx.x == 0) {
    float t = 0.f;
    for (int i = 0; i < 8; ++i) t += red[i];
    mean[ch] = t / (float)HW;
  }
}

// ============ Kernel 5: SE MLP gate (+ zero GroupNorm accumulators) ========
__global__ __launch_bounds__(128) void k_segate(const float* __restrict__ mean,
                                                const float* __restrict__ w1,
                                                const float* __restrict__ b1,
                                                const float* __restrict__ w2,
                                                const float* __restrict__ b2,
                                                float* __restrict__ gate,
                                                float* __restrict__ gstats) {
  __shared__ float hmid[32];
  const int t = threadIdx.x;
  if (t < 16) gstats[t] = 0.f;
  if (t < 32) {
    float a = b1[t];
    for (int c = 0; c < 128; ++c) a += w1[t * 128 + c] * mean[c];
    hmid[t] = a / (1.f + __expf(-a));  // silu
  }
  __syncthreads();
  float a = b2[t];
  for (int i = 0; i < 32; ++i) a += w2[t * 32 + i] * hmid[i];
  gate[t] = 1.f / (1.f + __expf(-a));
}

// ========== Kernel 6: output 1x1 conv (WMMA) + group-sum atomics ===========
__global__ __launch_bounds__(256) void k_outconv(const float* __restrict__ ao,
                                                 const float* __restrict__ gate,
                                                 const float* __restrict__ ow,
                                                 const float* __restrict__ ob,
                                                 float* __restrict__ y,
                                                 float* __restrict__ gstats) {
  __shared__ _Float16 xT[128 * 128];  // gated activations [p][c] f16
  const int lane = threadIdx.x & 31, wave = threadIdx.x >> 5;
  const int h = lane >> 4, lm = lane & 15;
  const int pbase = blockIdx.x * 128;

  for (int c = wave * 16; c < wave * 16 + 16; ++c) {
    const float g = gate[c];
    v4f xv = *(const v4f*)(ao + (size_t)c * HW + pbase + lane * 4);
#pragma unroll
    for (int u = 0; u < 4; ++u)
      xT[(lane * 4 + u) * 128 + c] = (_Float16)(xv[u] * g);
  }
  __syncthreads();

  v16h aw[4];
#pragma unroll
  for (int cb = 0; cb < 4; ++cb)
    aw[cb] = cvt_a(ow + (size_t)(wave * 16 + lm) * 128 + cb * 32, h);
  float bias[8];
#pragma unroll
  for (int r = 0; r < 8; ++r) bias[r] = ob[wave * 16 + r + 8 * h];

  float s1 = 0.f, s2 = 0.f;
  for (int pj = 0; pj < 8; ++pj) {
    v8f acc = {0, 0, 0, 0, 0, 0, 0, 0};
#pragma unroll
    for (int cb = 0; cb < 4; ++cb) {
      v16h b = load16(xT + (pj * 16 + lm) * 128 + cb * 32, h);
      acc = wmma16(aw[cb], b, acc);
    }
#pragma unroll
    for (int r = 0; r < 8; ++r) {
      const float yv = acc[r] + bias[r];
      y[(size_t)(wave * 16 + r + 8 * h) * HW + pbase + pj * 16 + lm] = yv;
      s1 += yv;
      s2 += yv * yv;
    }
  }
#pragma unroll
  for (int off = 16; off; off >>= 1) {
    s1 += __shfl_xor(s1, off, 32);
    s2 += __shfl_xor(s2, off, 32);
  }
  if (lane == 0) {           // wave `w` owns channel group `w` exactly
    atomicAdd(gstats + wave, s1);
    atomicAdd(gstats + 8 + wave, s2);
  }
}

// ===================== Kernel 7: GroupNorm finalize ========================
__global__ __launch_bounds__(256) void k_gn(const float* __restrict__ y,
                                            const float* __restrict__ gstats,
                                            const float* __restrict__ gw,
                                            const float* __restrict__ gb,
                                            float* __restrict__ out) {
  const size_t idx = (size_t)blockIdx.x * 256 + threadIdx.x;
  const int ch = (int)(idx >> 14);
  const int g = ch >> 4;
  const float invN = 1.0f / (16.0f * 16384.0f);
  const float mu = gstats[g] * invN;
  const float var = gstats[8 + g] * invN - mu * mu;
  const float rs = rsqrtf(var + GN_EPS);
  out[idx] = (y[idx] - mu) * rs * gw[ch] + gb[ch];
}

// ============================== Launcher ===================================
extern "C" void kernel_launch(void* const* d_in, const int* in_sizes, int n_in,
                              void* d_out, int out_size, void* d_ws, size_t ws_size,
                              hipStream_t stream) {
  const float* x  = (const float*)d_in[0];
  const float* wq = (const float*)d_in[1];
  const float* w1 = (const float*)d_in[2];
  const float* b1 = (const float*)d_in[3];
  const float* w2 = (const float*)d_in[4];
  const float* b2 = (const float*)d_in[5];
  const float* ow = (const float*)d_in[6];
  const float* ob = (const float*)d_in[7];
  const float* gw = (const float*)d_in[8];
  const float* gb = (const float*)d_in[9];
  float* out = (float*)d_out;

  char* ws = (char*)d_ws;
  const size_t MB = 1024 * 1024;
  _Float16* qh   = (_Float16*)(ws + 0);
  _Float16* kh   = (_Float16*)(ws + 4 * MB);
  _Float16* vh   = (_Float16*)(ws + 8 * MB);
  float*    mz   = (float*)(ws + 12 * MB);      // interleaved (max, 1/Z), 512 KB
  float*    ao   = (float*)(ws + 13 * MB);
  float*    smal = (float*)(ws + 21 * MB);
  float*    mean   = smal;
  float*    gate   = smal + 128;
  float*    gstats = smal + 256;  // 8 sums + 8 sumsqs
  float*    yb   = (float*)(ws + 0);  // reuse q/k region (dead after attn)

  k_qkv     <<<128,  256, 0, stream>>>(x, wq, qh, kh, vh);
  k_rowstats<<<512,  256, 0, stream>>>(qh, kh, mz);
  k_attnout <<<512,  256, 0, stream>>>(qh, kh, vh, mz, ao);
  k_chmean  <<<128,  256, 0, stream>>>(ao, mean);
  k_segate  <<<1,    128, 0, stream>>>(mean, w1, b1, w2, b2, gate, gstats);
  k_outconv <<<128,  256, 0, stream>>>(ao, gate, ow, ob, yb, gstats);
  k_gn      <<<8192, 256, 0, stream>>>(yb, gstats, gw, gb, out);
}
